// AttentionKVRM_68393059222052
// MI455X (gfx1250) — matrix operand order repaired and checked by reading, not compiled, not run on hardware
//
#include <hip/hip_runtime.h>
#include <math.h>

typedef __attribute__((ext_vector_type(16))) _Float16 v16h;
typedef __attribute__((ext_vector_type(8)))  float    v8f;

struct __align__(16) h8 { _Float16 h[8]; };

#define SEQ    2048
#define HDIM   128
#define NHEADS 16
#define NBATCH 4
#define NPAT   4
#define WPT    (SEQ/32)      /* 64 packed mask words per (pattern,t) row */
#define BM     64            /* query rows per block (4 waves x 16)      */
#define BN     64            /* key tile                                 */
#define NWAVE  4
#define NKT    (SEQ/BN)      /* 32 key tiles                             */
#define KPAD   (HDIM + 8)    /* 136 halves per ldsK row ([key][d]), 272B = 17*16 */
#define VPAD   (BN + 8)      /* 72 halves per ldsV row ([d][key]), 144B = 9*16   */

union U8F  { v8f  v; float    f[8];  };
union U16H { v16h v; _Float16 h[16]; unsigned u[8]; };

__device__ __forceinline__ v8f zero_v8f() {
  U8F z;
#pragma unroll
  for (int i = 0; i < 8; ++i) z.f[i] = 0.0f;
  return z.v;
}

// ---- CDNA5 async global->LDS copy (ASYNCcnt-tracked), 16B per lane ----
__device__ __forceinline__ void async_ld16(void* lds, const void* g) {
  unsigned      l = (unsigned)(uintptr_t)lds;            // low 32 bits = LDS offset
  unsigned long long a = (unsigned long long)(uintptr_t)g;
  asm volatile("global_load_async_to_lds_b128 %0, %1, off" :: "v"(l), "v"(a) : "memory");
}
__device__ __forceinline__ void wait_async0() {
#if __has_builtin(__builtin_amdgcn_s_wait_asynccnt)
  __builtin_amdgcn_s_wait_asynccnt(0);
#else
  asm volatile("s_wait_asynccnt 0" ::: "memory");
#endif
}

// ---------------------------------------------------------------------------
// Kernel 1: per-head pattern selector MLP (3 -> 32 -> 4) + argmax
// ---------------------------------------------------------------------------
__global__ void selector_kernel(const float* __restrict__ w1, const float* __restrict__ b1,
                                const float* __restrict__ w2, const float* __restrict__ b2,
                                int* __restrict__ pat_idx) {
  int h = threadIdx.x;
  if (h >= NHEADS) return;
  float f0 = 1.0f, f1 = (float)h / 12.0f, f2 = 0.5f;
  float logits[NPAT];
#pragma unroll
  for (int k = 0; k < NPAT; ++k) logits[k] = b2[k];
  for (int j = 0; j < 32; ++j) {
    float hij = f0 * w1[0*32 + j] + f1 * w1[1*32 + j] + f2 * w1[2*32 + j] + b1[j];
    hij = fmaxf(hij, 0.0f);
#pragma unroll
    for (int k = 0; k < NPAT; ++k) logits[k] += hij * w2[j*NPAT + k];
  }
  int best = 0; float bv = logits[0];
#pragma unroll
  for (int k = 1; k < NPAT; ++k) if (logits[k] > bv) { bv = logits[k]; best = k; }
  pat_idx[h] = best;
}

// ---------------------------------------------------------------------------
// Kernel 2: pack pattern[p][s][t] > 0 bits along s: packed[p][t][s/32]
// ---------------------------------------------------------------------------
__global__ void maskpack_kernel(const float* __restrict__ patterns, unsigned* __restrict__ packed) {
  __shared__ unsigned char tile[32][33];
  const int p  = blockIdx.z;
  const int s0 = blockIdx.x * 32;
  const int t0 = blockIdx.y * 32;
  const int tx = threadIdx.x, ty = threadIdx.y;
  float v = patterns[(size_t)p * SEQ * SEQ + (size_t)(s0 + ty) * SEQ + (t0 + tx)];
  tile[ty][tx] = (v > 0.0f) ? 1 : 0;           // sigmoid(x) > 0.5 <=> x > 0
  __syncthreads();
  int bit = tile[tx][ty];                       // lane tx indexes s, wave ty indexes t
#if __has_builtin(__builtin_amdgcn_ballot_w32)
  unsigned word = __builtin_amdgcn_ballot_w32(bit != 0);
  if (tx == 0)
    packed[(size_t)p * SEQ * WPT + (size_t)(t0 + ty) * WPT + (s0 >> 5)] = word;
#else
  if (tx == 0) {
    unsigned word = 0;
    for (int b = 0; b < 32; ++b) word |= ((unsigned)tile[b][ty]) << b;
    packed[(size_t)p * SEQ * WPT + (size_t)(t0 + ty) * WPT + (s0 >> 5)] = word;
  }
#endif
}

// ---------------------------------------------------------------------------
// Kernel 3a: fp32 -> f16 elementwise (8 elems / thread, b128 stores)
// ---------------------------------------------------------------------------
__global__ void cvt_f16_kernel(const float* __restrict__ src, _Float16* __restrict__ dst, int n8) {
  int i = blockIdx.x * blockDim.x + threadIdx.x;
  if (i >= n8) return;
  const float4* s4 = (const float4*)src + 2 * (size_t)i;
  float4 a = s4[0], b = s4[1];
  h8 o;
  o.h[0] = (_Float16)a.x; o.h[1] = (_Float16)a.y; o.h[2] = (_Float16)a.z; o.h[3] = (_Float16)a.w;
  o.h[4] = (_Float16)b.x; o.h[5] = (_Float16)b.y; o.h[6] = (_Float16)b.z; o.h[7] = (_Float16)b.w;
  *((h8*)dst + i) = o;
}

// ---------------------------------------------------------------------------
// Kernel 3b: V [bh][s][d] fp32 -> Vt [bh][d][s] f16 (32x32 LDS transpose)
// ---------------------------------------------------------------------------
__global__ void transpose_v_kernel(const float* __restrict__ V, _Float16* __restrict__ Vt) {
  __shared__ _Float16 tile[32][33];
  const int bh = blockIdx.z;
  const int s0 = blockIdx.x * 32;
  const int d0 = blockIdx.y * 32;
  const int tx = threadIdx.x, ty = threadIdx.y;
  tile[ty][tx] = (_Float16)V[((size_t)bh * SEQ + s0 + ty) * HDIM + d0 + tx];
  __syncthreads();
  Vt[((size_t)bh * HDIM + d0 + ty) * SEQ + s0 + tx] = tile[tx][ty];
}

// ---------------------------------------------------------------------------
// Kernel 4: flash attention with multiplicative binary mask.
// 128 threads = 4 waves; wave owns a 16-row M tile. PRE=true: f16 inputs +
// async double-buffered LDS tiles; PRE=false: fp32 inputs, convert on the fly.
// ---------------------------------------------------------------------------
template <bool PRE>
__launch_bounds__(128, 1)
__global__ void attn_kernel(const float* __restrict__ Qf, const float* __restrict__ Kf,
                            const float* __restrict__ Vf,
                            const _Float16* __restrict__ Qh, const _Float16* __restrict__ Kh,
                            const _Float16* __restrict__ Vth,
                            const unsigned* __restrict__ packed,
                            const int* __restrict__ pat_idx,
                            float* __restrict__ Out) {
  __shared__ __align__(16) _Float16 ldsK[2][BN * KPAD];    // K tile  [key][d]
  __shared__ __align__(16) _Float16 ldsV[2][HDIM * VPAD];  // V tile  [d][key]
  __shared__ __align__(16) _Float16 ldsP[NWAVE * 16 * BN]; // per-wave P buffer

  const int tid  = threadIdx.x;
  const int wave = tid >> 5;
  const int lane = tid & 31;
  const int half = lane >> 4;
  const int n16  = lane & 15;

  const int bh = blockIdx.x;                    // b*NHEADS + h
  const int h  = bh & (NHEADS - 1);
  const int m0 = blockIdx.y * BM + wave * 16;

  const int p = pat_idx[h];
  const unsigned* mbase = packed + (size_t)p * SEQ * WPT + (m0 >> 5);
  const int mshift = m0 & 31;

  // ---- Q fragments, A layout (lane: M = lane%16; interleaved K pattern) ----
  U16H qa[4];
  if constexpr (PRE) {
    const _Float16* qrow = Qh + ((size_t)bh * SEQ + (size_t)(m0 + n16)) * HDIM;
#pragma unroll
    for (int kc = 0; kc < 4; ++kc)
#pragma unroll
      for (int r = 0; r < 8; ++r) {
        int k = kc*32 + ((r < 4) ? 0 : 16) + 8*half + 2*(r & 3);
        qa[kc].u[r] = *(const unsigned*)(qrow + k);
      }
  } else {
    const float* qrow = Qf + ((size_t)bh * SEQ + (size_t)(m0 + n16)) * HDIM;
#pragma unroll
    for (int kc = 0; kc < 4; ++kc)
#pragma unroll
      for (int r = 0; r < 8; ++r) {
        int k = kc*32 + ((r < 4) ? 0 : 16) + 8*half + 2*(r & 3);
        qa[kc].h[2*r+0] = (_Float16)qrow[k+0];
        qa[kc].h[2*r+1] = (_Float16)qrow[k+1];
      }
  }

  U8F o[8];
#pragma unroll
  for (int nd = 0; nd < 8; ++nd) o[nd].v = zero_v8f();
  float rm[8], rl[8];
#pragma unroll
  for (int r = 0; r < 8; ++r) { rm[r] = -3.402823466e38f; rl[r] = 0.0f; }

  const float scale = 0.08838834764831845f;     // 1/sqrt(128)

  // ---- async prefetch of one K/V tile into LDS buffer `buf` (PRE only) ----
  auto prefetch = [&](int t0, int buf) {
#pragma unroll
    for (int i = 0; i < 8; ++i) {               // K tile: 1024 16B chunks
      int idx = i * 128 + tid;
      int key = idx >> 4, c8 = (idx & 15) << 3;
      async_ld16(ldsK[buf] + key * KPAD + c8,
                 Kh + ((size_t)bh * SEQ + t0 + key) * HDIM + c8);
    }
#pragma unroll
    for (int i = 0; i < 8; ++i) {               // V^T tile: 1024 16B chunks
      int idx = i * 128 + tid;
      int d = idx >> 3, c8 = (idx & 7) << 3;
      async_ld16(ldsV[buf] + d * VPAD + c8,
                 Vth + ((size_t)bh * HDIM + d) * SEQ + t0 + c8);
    }
  };

  if constexpr (PRE) prefetch(0, 0);

  for (int kt = 0; kt < NKT; ++kt) {
    const int t0  = kt * BN;
    const int buf = PRE ? (kt & 1) : 0;

    if constexpr (PRE) {
      wait_async0();            // my prefetch of `buf` done
      __syncthreads();          // everyone's prefetch done; prev compute done
      if (kt + 1 < NKT) prefetch(t0 + BN, buf ^ 1);
    } else {
      __syncthreads();
      const float* Kt = Kf + ((size_t)bh * SEQ + t0) * HDIM;
      const float* Vt = Vf + ((size_t)bh * SEQ + t0) * HDIM;
#pragma unroll
      for (int i = 0; i < (BN * HDIM / 4) / 128; ++i) {
        int idx = i * 128 + tid;
        int key = idx >> 5;
        int c4  = (idx & 31) << 2;
        const float4 kv = *(const float4*)(Kt + (size_t)key * HDIM + c4);
        _Float16* kd = ldsK[0] + key * KPAD + c4;
        kd[0] = (_Float16)kv.x; kd[1] = (_Float16)kv.y;
        kd[2] = (_Float16)kv.z; kd[3] = (_Float16)kv.w;
        const float4 vv = *(const float4*)(Vt + (size_t)key * HDIM + c4);
        ldsV[0][(c4+0)*VPAD + key] = (_Float16)vv.x;
        ldsV[0][(c4+1)*VPAD + key] = (_Float16)vv.y;
        ldsV[0][(c4+2)*VPAD + key] = (_Float16)vv.z;
        ldsV[0][(c4+3)*VPAD + key] = (_Float16)vv.w;
      }
      __syncthreads();
    }

    const _Float16* kb = ldsK[buf];
    const _Float16* vb = ldsV[buf];

    // ---- S = Q K^T (16 x 64): 16 WMMAs, two-deep fragment pipeline ----
    U8F s[4];
    {
      U16H f0, f1;
      auto ldb = [&](U16H& dst, int j) {        // j = nt*4 + kc
        int nt = j >> 2, kc = j & 3;
#pragma unroll
        for (int r = 0; r < 8; ++r)
          dst.u[r] = *(const unsigned*)(kb + (nt*16 + n16) * KPAD + kc*32 + ((r < 4) ? 0 : 16) + 8*half + 2*(r & 3));
      };
      ldb(f0, 0);
      v8f acc = zero_v8f();
#pragma unroll
      for (int j = 0; j < 16; ++j) {
        U16H& cur = (j & 1) ? f1 : f0;
        U16H& nxt = (j & 1) ? f0 : f1;
        if (j < 15) ldb(nxt, j + 1);
        acc = __builtin_amdgcn_wmma_f32_16x16x32_f16(false, qa[j & 3].v, false, cur.v,
                                                     (short)0, acc, false, false);
        if ((j & 3) == 3) { s[j >> 2].v = acc; acc = zero_v8f(); }
      }
    }

    // ---- scale, multiplicative binary mask, online softmax ----
    unsigned mw[4];
#pragma unroll
    for (int nt = 0; nt < 4; ++nt)
      mw[nt] = mbase[(size_t)(t0 + nt*16 + n16) * WPT];

    float tmax[8];
#pragma unroll
    for (int r = 0; r < 8; ++r) tmax[r] = -3.402823466e38f;
#pragma unroll
    for (int nt = 0; nt < 4; ++nt)
#pragma unroll
      for (int r = 0; r < 8; ++r) {
        float sv = s[nt].f[r] * scale;
        sv = ((mw[nt] >> (mshift + 8*half + r)) & 1u) ? sv : 0.0f;
        s[nt].f[r] = sv;
        tmax[r] = fmaxf(tmax[r], sv);
      }
#pragma unroll
    for (int off = 1; off < 16; off <<= 1)
#pragma unroll
      for (int r = 0; r < 8; ++r)
        tmax[r] = fmaxf(tmax[r], __shfl_xor(tmax[r], off, 32));

    float alpha[8], tsum[8];
#pragma unroll
    for (int r = 0; r < 8; ++r) {
      float mn = fmaxf(rm[r], tmax[r]);
      alpha[r] = __expf(rm[r] - mn);
      rm[r] = mn;
      tsum[r] = 0.0f;
    }
#pragma unroll
    for (int nt = 0; nt < 4; ++nt)
#pragma unroll
      for (int r = 0; r < 8; ++r) {
        float pv = __expf(s[nt].f[r] - rm[r]);
        s[nt].f[r] = pv;
        tsum[r] += pv;
      }
#pragma unroll
    for (int off = 1; off < 16; off <<= 1)
#pragma unroll
      for (int r = 0; r < 8; ++r) tsum[r] += __shfl_xor(tsum[r], off, 32);
#pragma unroll
    for (int r = 0; r < 8; ++r) rl[r] = rl[r] * alpha[r] + tsum[r];
#pragma unroll
    for (int nd = 0; nd < 8; ++nd)
#pragma unroll
      for (int r = 0; r < 8; ++r) o[nd].f[r] *= alpha[r];

    // ---- P: C-layout -> LDS (f16) -> A-layout fragments ----
    _Float16* pb = ldsP + wave * (16 * BN);
#pragma unroll
    for (int nt = 0; nt < 4; ++nt)
#pragma unroll
      for (int r = 0; r < 8; ++r)
        pb[(r + 8*half) * BN + nt*16 + n16] = (_Float16)s[nt].f[r];

    U16H pa[2];
#pragma unroll
    for (int kc = 0; kc < 2; ++kc)
#pragma unroll
      for (int r = 0; r < 8; ++r) {
        int k = kc*32 + ((r < 4) ? 0 : 16) + 8*half + 2*(r & 3);
        pa[kc].u[r] = *(const unsigned*)(pb + n16 * BN + k);
      }

    // ---- O += P x V (16 WMMAs), two-deep fragment pipeline ----
    {
      U16H g0, g1;
      auto ldv = [&](U16H& dst, int j) {        // j = nd*2 + kc
        int nd = j >> 1, kc = j & 1;
#pragma unroll
        for (int r = 0; r < 8; ++r)
          dst.u[r] = *(const unsigned*)(vb + (nd*16 + n16) * VPAD + kc*32 + ((r < 4) ? 0 : 16) + 8*half + 2*(r & 3));
      };
      ldv(g0, 0);
#pragma unroll
      for (int j = 0; j < 16; ++j) {
        U16H& cur = (j & 1) ? g1 : g0;
        U16H& nxt = (j & 1) ? g0 : g1;
        if (j < 15) ldv(nxt, j + 1);
        o[j >> 1].v = __builtin_amdgcn_wmma_f32_16x16x32_f16(false, pa[j & 1].v, false, cur.v,
                                                             (short)0, o[j >> 1].v, false, false);
      }
    }
  }

  // ---- epilogue: normalize by row sum and store ----
  float inv[8];
#pragma unroll
  for (int r = 0; r < 8; ++r) inv[r] = 1.0f / rl[r];
#pragma unroll
  for (int nd = 0; nd < 8; ++nd)
#pragma unroll
    for (int r = 0; r < 8; ++r) {
      int row = m0 + r + 8*half;
      int d   = nd*16 + n16;
      Out[((size_t)bh * SEQ + row) * HDIM + d] = o[nd].f[r] * inv[r];
    }
}

// ---------------------------------------------------------------------------
extern "C" void kernel_launch(void* const* d_in, const int* in_sizes, int n_in,
                              void* d_out, int out_size, void* d_ws, size_t ws_size,
                              hipStream_t stream) {
  const float* Q        = (const float*)d_in[0];
  const float* K        = (const float*)d_in[1];
  const float* V        = (const float*)d_in[2];
  const float* patterns = (const float*)d_in[3];
  const float* w1       = (const float*)d_in[4];
  const float* b1       = (const float*)d_in[5];
  const float* w2       = (const float*)d_in[6];
  const float* b2       = (const float*)d_in[7];
  float* Out            = (float*)d_out;

  const size_t nElem   = (size_t)NBATCH * NHEADS * SEQ * HDIM;  // 16.78M
  const size_t szPack  = (size_t)NPAT * SEQ * WPT * sizeof(unsigned); // 2 MB
  const size_t szHalf  = nElem * sizeof(_Float16);              // 32 MB

  char* ws = (char*)d_ws;
  int*       pat_idx = (int*)ws;
  unsigned*  packed  = (unsigned*)(ws + 4096);
  _Float16*  Qh      = (_Float16*)(ws + 4096 + szPack);
  _Float16*  Kh      = Qh + nElem;
  _Float16*  Vth     = Kh + nElem;
  const size_t needFull = 4096 + szPack + 3 * szHalf;
  const bool pre = (ws_size >= needFull);

  selector_kernel<<<1, 32, 0, stream>>>(w1, b1, w2, b2, pat_idx);
  maskpack_kernel<<<dim3(SEQ/32, SEQ/32, NPAT), dim3(32, 32), 0, stream>>>(patterns, packed);

  if (pre) {
    const int n8 = (int)(nElem / 8);
    cvt_f16_kernel<<<(n8 + 255) / 256, 256, 0, stream>>>(Q, Qh, n8);
    cvt_f16_kernel<<<(n8 + 255) / 256, 256, 0, stream>>>(K, Kh, n8);
    transpose_v_kernel<<<dim3(SEQ/32, HDIM/32, NBATCH*NHEADS), dim3(32, 32), 0, stream>>>(V, Vth);
    attn_kernel<true><<<dim3(NBATCH * NHEADS, SEQ / BM), 128, 0, stream>>>(
        Q, K, V, Qh, Kh, Vth, packed, pat_idx, Out);
  } else {
    attn_kernel<false><<<dim3(NBATCH * NHEADS, SEQ / BM), 128, 0, stream>>>(
        Q, K, V, Qh, Kh, Vth, packed, pat_idx, Out);
  }
}
